// MultiHeadAttention_75960791597788
// MI455X (gfx1250) — matrix operand order repaired and checked by reading, not compiled, not run on hardware
//
#include <hip/hip_runtime.h>
#include <hip/hip_bf16.h>

typedef _Float16 half_t;
typedef __attribute__((ext_vector_type(4)))  _Float16 v4h;
typedef __attribute__((ext_vector_type(8)))  _Float16 v8h;
typedef __attribute__((ext_vector_type(16))) _Float16 v16h;
typedef __attribute__((ext_vector_type(8)))  float    v8f;

#define NB  4
#define SEQ 2048
#define EMB 1024
#define NH  16
#define DHD 64

// ---------------------------------------------------------------------------
// WMMA helpers (CDNA5 gfx1250, wave32). D = A(16x32 f16) x B(32x16 f16) + C(f32)
// ---------------------------------------------------------------------------
__device__ __forceinline__ v8f wmma_f16(v16h a, v16h b, v8f c) {
  return __builtin_amdgcn_wmma_f32_16x16x32_f16(
      /*neg_a=*/false, a, /*neg_b=*/false, b,
      /*c_mod=*/(short)0, c, /*reuse_a=*/false, /*reuse_b=*/false);
}

// A fragment: lane L holds row (L&15); K elems {(L>>4)*8 + 0..7} and {+16..+23}
__device__ __forceinline__ v16h frag_a(const half_t* tile, int ldm, int lane) {
  const int r = lane & 15, g = lane >> 4;
  const v8h* lo = (const v8h*)(tile + (size_t)r * ldm + g * 8);
  const v8h* hi = (const v8h*)(tile + (size_t)r * ldm + g * 8 + 16);
  return __builtin_shufflevector(*lo, *hi, 0,1,2,3,4,5,6,7,8,9,10,11,12,13,14,15);
}

// B fragment: lane L holds column (L&15); K elems {(L>>4)*8 + 0..7} and {+16..+23}
__device__ __forceinline__ v16h frag_b(const half_t* tile, int ldm, int lane) {
  const int c = lane & 15, g = lane >> 4;
  const v8h* lo = (const v8h*)(tile + (size_t)c * ldm + g * 8);
  const v8h* hi = (const v8h*)(tile + (size_t)c * ldm + g * 8 + 16);
  return __builtin_shufflevector(*lo, *hi, 0,1,2,3,4,5,6,7,8,9,10,11,12,13,14,15);
}

// ---------------------------------------------------------------------------
// GEMM: Out[M x 1024] = X[M x 1024] @ W[1024 x 1024]^T + bias   (NT form)
// Block tile 128x128, 8 waves, each wave 32x64 (2x4 WMMA tiles), K staged 64.
// Register-level software pipelining: global fetch of tile t+1 overlaps the
// WMMA compute of tile t (single LDS buffer, waits sunk to next ds_store).
// XT = float (convert to f16 while staging) or half_t. OT = half_t or float.
// ---------------------------------------------------------------------------
template <typename XT, typename OT>
__global__ __launch_bounds__(256)
void gemm_xwt_bias(const XT* __restrict__ X, const float* __restrict__ W,
                   const float* __restrict__ bias, OT* __restrict__ Out) {
  constexpr int BM = 128, BN = 128, BK = 64, LDT = BK + 8;
  __shared__ half_t As[BM][LDT];
  __shared__ half_t Bs[BN][LDT];

  const int tid = threadIdx.x, lane = tid & 31, w = tid >> 5;
  const int wm = w >> 1, wn = w & 1;           // 4 waves along M, 2 along N
  const int m0 = blockIdx.y * BM, n0 = blockIdx.x * BN;
  const int row = tid >> 1, seg = tid & 1;     // 2 threads per staged row

  v8f acc[2][4] = {};

  // prefetch registers for the next K tile
  float4 xa_f[8];
  v8h    xa_h[4];
  float4 wb[8];

  auto preload = [&](int kt) {
    if constexpr (sizeof(XT) == 4) {
      const float4* src = (const float4*)(X + (size_t)(m0 + row) * EMB + kt + seg * 32);
      #pragma unroll
      for (int i = 0; i < 8; ++i) xa_f[i] = src[i];
    } else {
      const v8h* src = (const v8h*)(X + (size_t)(m0 + row) * EMB + kt + seg * 32);
      #pragma unroll
      for (int i = 0; i < 4; ++i) xa_h[i] = src[i];
    }
    const float4* srcw = (const float4*)(W + (size_t)(n0 + row) * EMB + kt + seg * 32);
    #pragma unroll
    for (int i = 0; i < 8; ++i) wb[i] = srcw[i];
  };

  auto stage = [&]() {
    if constexpr (sizeof(XT) == 4) {
      #pragma unroll
      for (int i = 0; i < 8; ++i) {
        float4 f = xa_f[i];
        v4h h = { (half_t)f.x, (half_t)f.y, (half_t)f.z, (half_t)f.w };
        *(v4h*)&As[row][seg * 32 + i * 4] = h;
      }
    } else {
      #pragma unroll
      for (int i = 0; i < 4; ++i) *(v8h*)&As[row][seg * 32 + i * 8] = xa_h[i];
    }
    #pragma unroll
    for (int i = 0; i < 8; ++i) {
      float4 f = wb[i];
      v4h h = { (half_t)f.x, (half_t)f.y, (half_t)f.z, (half_t)f.w };
      *(v4h*)&Bs[row][seg * 32 + i * 4] = h;
    }
  };

  preload(0);

  for (int kt = 0; kt < EMB; kt += BK) {
    __syncthreads();            // previous tile's compute finished reading LDS
    stage();                    // regs -> LDS (implicit wait on global loads)
    __syncthreads();
    if (kt + BK < EMB) preload(kt + BK);   // issue next fetch, no wait here

    #pragma unroll
    for (int ks = 0; ks < BK / 32; ++ks) {
      v16h a[2], b[4];
      #pragma unroll
      for (int mt = 0; mt < 2; ++mt)
        a[mt] = frag_a(&As[wm * 32 + mt * 16][ks * 32], LDT, lane);
      #pragma unroll
      for (int nt = 0; nt < 4; ++nt)
        b[nt] = frag_b(&Bs[wn * 64 + nt * 16][ks * 32], LDT, lane);
      #pragma unroll
      for (int mt = 0; mt < 2; ++mt)
        #pragma unroll
        for (int nt = 0; nt < 4; ++nt)
          acc[mt][nt] = wmma_f16(a[mt], b[nt], acc[mt][nt]);
    }
  }

  // ---- epilogue: C layout lane=(N), VGPR v = row v + 8*(lane>=16) ----
  const int half8 = (lane >> 4) * 8;
  const int cl = lane & 15;
  #pragma unroll
  for (int mt = 0; mt < 2; ++mt) {
    #pragma unroll
    for (int nt = 0; nt < 4; ++nt) {
      const int col = n0 + wn * 64 + nt * 16 + cl;
      const float bv = bias[col];
      #pragma unroll
      for (int v = 0; v < 8; ++v) {
        const int r = m0 + wm * 32 + mt * 16 + v + half8;
        Out[(size_t)r * EMB + col] = (OT)(acc[mt][nt][v] + bv);
      }
    }
  }
}

// ---------------------------------------------------------------------------
// Flash attention: one block = 128 query rows of one (n,h). 8 waves x 16 rows.
// Q pre-scaled by 1/8 (= 1/sqrt(DH), exact in f16). Streams 128-key chunks
// with register-level prefetch of the next K/V chunk behind compute.
// ---------------------------------------------------------------------------
__global__ __launch_bounds__(256)
void attention_kernel(const half_t* __restrict__ Q, const half_t* __restrict__ K,
                      const half_t* __restrict__ V, half_t* __restrict__ Y) {
  constexpr int LQ = DHD + 8;   // 72
  constexpr int LP = 128 + 8;   // 136
  __shared__ half_t Qs[128][LQ];
  __shared__ half_t Ks[128][LQ];
  __shared__ half_t Vt[DHD][LP];       // V transposed: Vt[d][t]
  __shared__ half_t Ps[8][16][LP];     // per-wave P tile (16 rows x 128 keys)

  const int tid = threadIdx.x, lane = tid & 31, w = tid >> 5;
  const int sblk = blockIdx.x;               // query block 0..15
  const int nh = blockIdx.y;                 // 0..63
  const int n = nh >> 4, h = nh & 15;
  const int row = tid >> 1, seg = tid & 1;
  const int half8 = (lane >> 4) * 8;
  const int cl = lane & 15;

  // ---- stage Q tile once, scaled by 0.125 ----
  {
    const v8h* src = (const v8h*)(Q + ((size_t)(n * SEQ + sblk * 128 + row)) * EMB
                                  + h * DHD + seg * 32);
    #pragma unroll
    for (int i = 0; i < 2; ++i) {
      v8h q = src[i];
      #pragma unroll
      for (int j = 0; j < 8; ++j) q[j] = (half_t)(q[j] * (half_t)0.125f);
      *(v8h*)&Qs[row][seg * 32 + i * 8] = q;
    }
  }
  __syncthreads();

  v16h aq[2];
  #pragma unroll
  for (int ks = 0; ks < 2; ++ks) aq[ks] = frag_a(&Qs[w * 16][ks * 32], LQ, lane);

  float m[8], l[8];
  v8f o[4] = {};
  #pragma unroll
  for (int v = 0; v < 8; ++v) { m[v] = -3.0e38f; l[v] = 0.0f; }

  const size_t kvbase = ((size_t)n * SEQ) * EMB + (size_t)h * DHD;

  // K/V chunk prefetch registers
  v8h kreg[2], vreg[2];
  auto preload_kv = [&](int tc) {
    const v8h* srck = (const v8h*)(K + kvbase + (size_t)(tc * 128 + row) * EMB + seg * 32);
    kreg[0] = srck[0]; kreg[1] = srck[1];
    const v8h* srcv = (const v8h*)(V + kvbase + (size_t)(tc * 128 + row) * EMB + seg * 32);
    vreg[0] = srcv[0]; vreg[1] = srcv[1];
  };

  preload_kv(0);

  for (int tc = 0; tc < SEQ / 128; ++tc) {
    __syncthreads();            // previous chunk's compute finished
    // ---- stage K chunk (row-major) and V chunk (transposed) from regs ----
    {
      #pragma unroll
      for (int i = 0; i < 2; ++i) *(v8h*)&Ks[row][seg * 32 + i * 8] = kreg[i];
      #pragma unroll
      for (int i = 0; i < 2; ++i) {
        v8h vv = vreg[i];
        #pragma unroll
        for (int j = 0; j < 8; ++j) Vt[seg * 32 + i * 8 + j][row] = vv[j];
      }
    }
    __syncthreads();
    if (tc + 1 < SEQ / 128) preload_kv(tc + 1);   // overlap with compute below

    // ---- scores: 16 query rows x 128 keys, K-dim = 64 (2 wmma steps) ----
    v8f sacc[8];
    #pragma unroll
    for (int nt = 0; nt < 8; ++nt) {
      v8f z = {};
      sacc[nt] = z;
      #pragma unroll
      for (int ks = 0; ks < 2; ++ks) {
        v16h b = frag_b(&Ks[nt * 16][ks * 32], LQ, lane);
        sacc[nt] = wmma_f16(aq[ks], b, sacc[nt]);
      }
    }

    // ---- online softmax: row max across 8 tiles + 16 lanes of half-wave ----
    float mc[8];
    #pragma unroll
    for (int v = 0; v < 8; ++v) {
      float x = sacc[0][v];
      #pragma unroll
      for (int nt = 1; nt < 8; ++nt) x = fmaxf(x, sacc[nt][v]);
      #pragma unroll
      for (int off = 1; off < 16; off <<= 1) x = fmaxf(x, __shfl_xor(x, off, 32));
      mc[v] = x;
    }

    float alpha[8], mn[8], psum[8];
    #pragma unroll
    for (int v = 0; v < 8; ++v) {
      mn[v]    = fmaxf(m[v], mc[v]);
      alpha[v] = __expf(m[v] - mn[v]);
      m[v]     = mn[v];
      psum[v]  = 0.0f;
    }
    #pragma unroll
    for (int dt = 0; dt < 4; ++dt)
      #pragma unroll
      for (int v = 0; v < 8; ++v) o[dt][v] *= alpha[v];

    // ---- P = exp(S - m), accumulate row sums, spill P (f16) to wave LDS ----
    #pragma unroll
    for (int nt = 0; nt < 8; ++nt) {
      #pragma unroll
      for (int v = 0; v < 8; ++v) {
        float p = __expf(sacc[nt][v] - mn[v]);
        psum[v] += p;
        Ps[w][v + half8][nt * 16 + cl] = (half_t)p;
      }
    }
    #pragma unroll
    for (int v = 0; v < 8; ++v) {
      float s = psum[v];
      #pragma unroll
      for (int off = 1; off < 16; off <<= 1) s += __shfl_xor(s, off, 32);
      l[v] = l[v] * alpha[v] + s;
    }

    // wave-local LDS turnaround: P stores must land before fragment reloads
    asm volatile("s_wait_dscnt 0x0" ::: "memory");

    // ---- O += P(16x128) x V(128x64): 4 k-steps x 4 d-tiles ----
    #pragma unroll
    for (int ks = 0; ks < 4; ++ks) {
      v16h pa = frag_a(&Ps[w][0][ks * 32], LP, lane);
      #pragma unroll
      for (int dt = 0; dt < 4; ++dt) {
        v16h vb = frag_b(&Vt[dt * 16][ks * 32], LP, lane);
        o[dt] = wmma_f16(pa, vb, o[dt]);
      }
    }
  }

  // ---- normalize and write Y[n][s][h*64+d] (f16) ----
  float inv_l[8];
  #pragma unroll
  for (int v = 0; v < 8; ++v) inv_l[v] = 1.0f / l[v];
  #pragma unroll
  for (int dt = 0; dt < 4; ++dt) {
    #pragma unroll
    for (int v = 0; v < 8; ++v) {
      const int s = sblk * 128 + w * 16 + v + half8;
      const int d = dt * 16 + cl;
      Y[((size_t)(n * SEQ + s)) * EMB + h * DHD + d] = (half_t)(o[dt][v] * inv_l[v]);
    }
  }
}

// ---------------------------------------------------------------------------
extern "C" void kernel_launch(void* const* d_in, const int* in_sizes, int n_in,
                              void* d_out, int out_size, void* d_ws, size_t ws_size,
                              hipStream_t stream) {
  (void)in_sizes; (void)n_in; (void)out_size; (void)ws_size;
  const float* query = (const float*)d_in[0];
  const float* key_  = (const float*)d_in[1];
  const float* value = (const float*)d_in[2];
  const float* Wq = (const float*)d_in[3];
  const float* bq = (const float*)d_in[4];
  const float* Wk = (const float*)d_in[5];
  const float* bk = (const float*)d_in[6];
  const float* Wv = (const float*)d_in[7];
  const float* bv = (const float*)d_in[8];
  const float* Wp = (const float*)d_in[9];
  const float* bp = (const float*)d_in[10];
  float* out = (float*)d_out;

  const size_t mat = (size_t)NB * SEQ * EMB;   // 8Mi elements
  half_t* Qh = (half_t*)d_ws;
  half_t* Kh = Qh + mat;
  half_t* Vh = Kh + mat;
  half_t* Yh = Vh + mat;                       // 64 MB total f16 workspace

  dim3 blk(256);
  dim3 gproj(EMB / 128, (NB * SEQ) / 128);     // (8, 64)
  gemm_xwt_bias<float, half_t><<<gproj, blk, 0, stream>>>(query, Wq, bq, Qh);
  gemm_xwt_bias<float, half_t><<<gproj, blk, 0, stream>>>(key_,  Wk, bk, Kh);
  gemm_xwt_bias<float, half_t><<<gproj, blk, 0, stream>>>(value, Wv, bv, Vh);

  dim3 gattn(SEQ / 128, NB * NH);              // (16, 64)
  attention_kernel<<<gattn, blk, 0, stream>>>(Qh, Kh, Vh, Yh);

  gemm_xwt_bias<half_t, float><<<gproj, blk, 0, stream>>>(Yh, Wp, bp, out);
}
